// BasicEdgeModel_49976239456345
// MI455X (gfx1250) — matrix operand order repaired and checked by reading, not compiled, not run on hardware
//
#include <hip/hip_runtime.h>

// CDNA5 (gfx1250) edge-MLP:  out = relu(concat(x[src],x[tgt],ea) @ W1 + b1) @ W2 + b2
// wave32; v_wmma_f32_16x16x32_f16 for both GEMMs, f32 accumulate.
// Round 2 -> 3: coalesced weight preload (col fastest across lanes), TPW 4->8,
// single-instruction ReLU. Structure (rolled k-loops, LDS-staged activations,
// aliased merged/h region) unchanged -- it compiled to spill-free WMMA pipelines.

typedef __attribute__((ext_vector_type(16))) _Float16 v16h;
typedef __attribute__((ext_vector_type(8)))  float    v8f;

#define N_EDGES   1600000
#define NODE_DIM  64
#define EDGE_DIM  32
#define HIDDEN    128
#define OUT_DIM   64

#define N_TILES        (N_EDGES / 16)                   // 100000
#define WAVES          8
#define TILES_PER_WAVE 8
#define TILES_PER_BLOCK (WAVES * TILES_PER_WAVE)        // 64
#define NBLOCKS ((N_TILES + TILES_PER_BLOCK - 1) / TILES_PER_BLOCK)  // 1563

#define KT1 5   // 160 / 32 k-tiles, GEMM1
#define NT1 8   // 128 / 16 n-tiles, GEMM1
#define KT2 4   // 128 / 32 k-tiles, GEMM2
#define NT2 4   //  64 / 16 n-tiles, GEMM2

#define MROW 84 // merged row pitch in dwords (160 halfs = 80 dwords, +4 pad: bank spread)
#define HROW 68 // h row pitch in dwords (128 halfs = 64 dwords, +4 pad)
#define MB_DWORDS (16 * MROW)   // 1344 dwords per wave (merged; h aliases this)

static __device__ __forceinline__ unsigned pack2h(float f0, float f1) {
    unsigned short u0 = __builtin_bit_cast(unsigned short, (_Float16)f0);
    unsigned short u1 = __builtin_bit_cast(unsigned short, (_Float16)f1);
    return (unsigned)u0 | ((unsigned)u1 << 16);
}

__global__ __launch_bounds__(256)
void edge_mlp_wmma(const float* __restrict__ x,
                   const float* __restrict__ edge_attr,
                   const float* __restrict__ W1,
                   const float* __restrict__ b1,
                   const float* __restrict__ W2,
                   const float* __restrict__ b2,
                   const int*   __restrict__ edge_index,
                   float*       __restrict__ out)
{
    // B-fragments pre-swizzled for v_wmma_f32_16x16x32_f16, lane-major:
    // dword index = frag*256 + l*8 + j ; V_j holds K = koff+2j+8*(j>>2) (lo), +1 (hi),
    // koff = (l>=16) ? 8 : 0, N = l & 15.
    __shared__ __align__(16) unsigned W1s[KT1 * NT1 * 256];     // 40 KB
    __shared__ __align__(16) unsigned W2s[KT2 * NT2 * 256];     // 16 KB
    __shared__ float b1s[HIDDEN];
    __shared__ float b2s[OUT_DIM];
    __shared__ __align__(16) unsigned mbuf[WAVES][MB_DWORDS];   // 43 KB: merged, then h (aliased)

    const int tid  = threadIdx.x;
    const int lane = tid & 31;
    const int wave = tid >> 5;
    const int lo   = lane & 15;       // N (and edge-row) index within half-wave
    const int hi   = lane >> 4;       // which K-half this lane supplies

    // ------- weight preload + f32->f16 swizzle (coalesced: col fastest) -------
    {
        const int j    = tid >> 5;            // K-pair index 0..7
        const int l    = tid & 31;            // lane slot
        const int n    = l & 15;
        const int koff = (l >> 4) * 8;
        for (int frag = 0; frag < KT1 * NT1; ++frag) {
            int kt = frag >> 3, nt = frag & 7;
            int k0  = kt * 32 + koff + 2 * j + (j >> 2) * 8;
            int col = nt * 16 + n;            // consecutive across 16 lanes -> 64B lines
            W1s[frag * 256 + l * 8 + j] =
                pack2h(W1[k0 * HIDDEN + col], W1[(k0 + 1) * HIDDEN + col]);
        }
        for (int frag = 0; frag < KT2 * NT2; ++frag) {
            int kt = frag >> 2, nt = frag & 3;
            int k0  = kt * 32 + koff + 2 * j + (j >> 2) * 8;
            int col = nt * 16 + n;
            W2s[frag * 256 + l * 8 + j] =
                pack2h(W2[k0 * OUT_DIM + col], W2[(k0 + 1) * OUT_DIM + col]);
        }
    }
    if (tid < HIDDEN)  b1s[tid] = b1[tid];
    if (tid < OUT_DIM) b2s[tid] = b2[tid];
    __syncthreads();

    unsigned* const mb = mbuf[wave];            // merged: [16][MROW] dwords
    _Float16* const hb = (_Float16*)mbuf[wave]; // h:      [16][2*HROW] halfs

    float bv1[NT1], bv2[NT2];
#pragma unroll
    for (int nt = 0; nt < NT1; ++nt) bv1[nt] = b1s[nt * 16 + lo];
#pragma unroll
    for (int nt = 0; nt < NT2; ++nt) bv2[nt] = b2s[nt * 16 + lo];

    // ---------------- per-wave tiles of 16 edges ----------------
    for (int it = 0; it < TILES_PER_WAVE; ++it) {
        const int tile = (blockIdx.x * WAVES + wave) * TILES_PER_WAVE + it;
        if (tile >= N_TILES) break;               // tail guard (wave-uniform)
        const int ebase = tile * 16;
        const int e     = ebase + lo;             // this lane's edge row (M)
        const int src   = edge_index[e];
        const int tgt   = edge_index[N_EDGES + e];

        // ---- stage merged row (f32 -> f16) into LDS: row lo, cols split by hi ----
        {
            const float4* s4 = (const float4*)(x + (long)src * NODE_DIM) + hi * 8;
            const float4* t4 = (const float4*)(x + (long)tgt * NODE_DIM) + hi * 8;
            const float4* a4 = (const float4*)(edge_attr + (long)e * EDGE_DIM) + hi * 4;
            unsigned* row = mb + lo * MROW;
#pragma unroll
            for (int i = 0; i < 8; ++i) {             // x[src]: cols hi*32 + 4i
                float4 v = s4[i];
                uint2 d; d.x = pack2h(v.x, v.y); d.y = pack2h(v.z, v.w);
                *(uint2*)(row + hi * 16 + 2 * i) = d;
            }
#pragma unroll
            for (int i = 0; i < 8; ++i) {             // x[tgt]: cols 64 + hi*32 + 4i
                float4 v = t4[i];
                uint2 d; d.x = pack2h(v.x, v.y); d.y = pack2h(v.z, v.w);
                *(uint2*)(row + 32 + hi * 16 + 2 * i) = d;
            }
#pragma unroll
            for (int i = 0; i < 4; ++i) {             // edge_attr: cols 128 + hi*16 + 4i
                float4 v = a4[i];
                uint2 d; d.x = pack2h(v.x, v.y); d.y = pack2h(v.z, v.w);
                *(uint2*)(row + 64 + hi * 8 + 2 * i) = d;
            }
        }

        // ---- GEMM1: [16,160] @ [160,128], accumulators seeded with bias ----
        v8f acc[NT1];
#pragma unroll
        for (int nt = 0; nt < NT1; ++nt)
#pragma unroll
            for (int r = 0; r < 8; ++r) acc[nt][r] = bv1[nt];

#pragma unroll 1
        for (int kt = 0; kt < KT1; ++kt) {
            const unsigned* ab = mb + lo * MROW + kt * 16 + hi * 4;
            union { uint4 v[2]; v16h h; } A;
            A.v[0] = *(const uint4*)ab;               // cols kt*32+hi*8 + 0..7
            A.v[1] = *(const uint4*)(ab + 8);         // cols kt*32+hi*8 + 16..23
            const unsigned* wb = W1s + kt * (NT1 * 256) + lane * 8;
#pragma unroll
            for (int nt = 0; nt < NT1; ++nt) {
                union { uint4 v[2]; v16h h; } B;
                B.v[0] = *(const uint4*)(wb + nt * 256);
                B.v[1] = *(const uint4*)(wb + nt * 256 + 4);
                acc[nt] = __builtin_amdgcn_wmma_f32_16x16x32_f16(
                    false, A.h, false, B.h, (short)0, acc[nt], false, false);
            }
        }

        // ---- ReLU + restage h as f16 into the (now dead) merged region ----
        // C frag: element r of lane = (M = r + 8*hi, N = nt*16 + lo)
#pragma unroll
        for (int nt = 0; nt < NT1; ++nt)
#pragma unroll
            for (int r = 0; r < 8; ++r) {
                hb[(r + 8 * hi) * (2 * HROW) + nt * 16 + lo] =
                    (_Float16)fmaxf(acc[nt][r], 0.0f);
            }

        // ---- GEMM2: [16,128] @ [128,64], accumulators seeded with bias ----
        v8f acc2[NT2];
#pragma unroll
        for (int nt = 0; nt < NT2; ++nt)
#pragma unroll
            for (int r = 0; r < 8; ++r) acc2[nt][r] = bv2[nt];

#pragma unroll 1
        for (int kt = 0; kt < KT2; ++kt) {
            const unsigned* ab = mb + lo * HROW + kt * 16 + hi * 4;
            union { uint4 v[2]; v16h h; } A;
            A.v[0] = *(const uint4*)ab;
            A.v[1] = *(const uint4*)(ab + 8);
            const unsigned* wb = W2s + kt * (NT2 * 256) + lane * 8;
#pragma unroll
            for (int nt = 0; nt < NT2; ++nt) {
                union { uint4 v[2]; v16h h; } B;
                B.v[0] = *(const uint4*)(wb + nt * 256);
                B.v[1] = *(const uint4*)(wb + nt * 256 + 4);
                acc2[nt] = __builtin_amdgcn_wmma_f32_16x16x32_f16(
                    false, A.h, false, B.h, (short)0, acc2[nt], false, false);
            }
        }

        // ---- write out [16,64] f32 (half-wave rows are 64B-contiguous) ----
#pragma unroll
        for (int nt = 0; nt < NT2; ++nt)
#pragma unroll
            for (int r = 0; r < 8; ++r) {
                int m = r + 8 * hi;
                out[(long)(ebase + m) * OUT_DIM + nt * 16 + lo] = acc2[nt][r];
            }
    }
}

extern "C" void kernel_launch(void* const* d_in, const int* in_sizes, int n_in,
                              void* d_out, int out_size, void* d_ws, size_t ws_size,
                              hipStream_t stream) {
    const float* x   = (const float*)d_in[0];
    const float* ea  = (const float*)d_in[1];
    const float* W1  = (const float*)d_in[2];
    const float* b1  = (const float*)d_in[3];
    const float* W2  = (const float*)d_in[4];
    const float* b2  = (const float*)d_in[5];
    const int*   ei  = (const int*)d_in[6];
    float*       out = (float*)d_out;
    (void)in_sizes; (void)n_in; (void)out_size; (void)d_ws; (void)ws_size;

    edge_mlp_wmma<<<NBLOCKS, 256, 0, stream>>>(x, ea, W1, b1, W2, b2, ei, out);
}
